// PointerGenerator_3135326126147
// MI455X (gfx1250) — matrix operand order repaired and checked by reading, not compiled, not run on hardware
//
#include <hip/hip_runtime.h>

#define CV 32000
#define CD 256
#define CH 512
#define CB 16
#define CS 400
#define CT 100
#define CNE 50
#define G4H 2048
#define VE (CV + CNE)

typedef __attribute__((ext_vector_type(16))) __bf16 v16bf;
typedef __attribute__((ext_vector_type(8)))  float  v8f;
typedef __attribute__((ext_vector_type(8)))  unsigned int v8u;
typedef __attribute__((ext_vector_type(4)))  unsigned int v4u;

#define WMMA_BF16(A,B,C) __builtin_amdgcn_wmma_f32_16x16x32_bf16(false,(A),false,(B),(short)0,(C),false,false)

__device__ __forceinline__ float sigm(float x){ return 1.f/(1.f+expf(-x)); }

// A fragment: 16x32 (MxK) bf16, row-major source with leading dim lda.
// lane<16 holds row m=lane, K = {0..7, 16..23}; lane>=16 holds row m=lane-16, K = {8..15, 24..31}.
__device__ __forceinline__ v16bf load_a(const __bf16* A, int lda, int k0, int lane){
  int m = lane & 15, hi = (lane >> 4) & 1;
  const __bf16* p = A + (size_t)m * lda + k0 + hi * 8;
  v4u lo = *(const v4u*)p;
  v4u hv = *(const v4u*)(p + 16);
  v8u r; r[0]=lo[0]; r[1]=lo[1]; r[2]=lo[2]; r[3]=lo[3];
         r[4]=hv[0]; r[5]=hv[1]; r[6]=hv[2]; r[7]=hv[3];
  return __builtin_bit_cast(v16bf, r);
}

// B fragment for Y = X @ W.T : B[k][n] = W[n][k]; W row-major (N x ldw) bf16.
// lane<16: K = {0..7, 16..23}; lane>=16: K = {8..15, 24..31}; lanes give N; kb = 8*(lane>>4).
__device__ __forceinline__ v16bf load_bT(const __bf16* Wm, int ldw, int n0, int k0, int lane){
  int nn = lane & 15, kb = ((lane >> 4) & 1) << 3;
  const __bf16* p = Wm + (size_t)(n0 + nn) * ldw + k0 + kb;
  v4u lo = *(const v4u*)p;
  v4u hv = *(const v4u*)(p + 16);
  v8u r; r[0]=lo[0]; r[1]=lo[1]; r[2]=lo[2]; r[3]=lo[3];
         r[4]=hv[0]; r[5]=hv[1]; r[6]=hv[2]; r[7]=hv[3];
  return __builtin_bit_cast(v16bf, r);
}

// ---------------- elementwise / gather kernels ----------------

__global__ void k_f2b(const float* __restrict__ s, __bf16* __restrict__ d, long n){
  long i = (long)blockIdx.x*256 + threadIdx.x;
  if (i < n) d[i] = (__bf16)s[i];
}

__global__ void k_embed_src(const int* __restrict__ src, const float* __restrict__ emb,
                            __bf16* __restrict__ xs){
  long i = (long)blockIdx.x*256 + threadIdx.x;
  if (i >= (long)CS*CB*CD) return;
  int d = (int)(i % CD); long r = i / CD; int b = (int)(r % CB); int s = (int)(r / CB);
  xs[i] = (__bf16)emb[(size_t)src[b*CS + s]*CD + d];
}

__global__ void k_embed_tgt(const int* __restrict__ tgt, const float* __restrict__ emb,
                            float* __restrict__ wt){
  long i = (long)blockIdx.x*256 + threadIdx.x;
  if (i >= (long)CT*CB*CD) return;
  int d = (int)(i % CD); long r = i / CD; int b = (int)(r % CB); int t = (int)(r / CB);
  wt[i] = emb[(size_t)tgt[b*CT + t]*CD + d];
}

// ---------------- generic WMMA GEMM: Y = X @ W.T (+bias)(+tanh) ----------------

__global__ __launch_bounds__(256) void k_gemm_xt(
    const __bf16* __restrict__ X, const __bf16* __restrict__ W,
    const float* __restrict__ bias, float* __restrict__ Yf,
    __bf16* __restrict__ Ybf, int M, int N, int K, int act)
{
  int wv = blockIdx.x * 8 + (threadIdx.x >> 5);
  int lane = threadIdx.x & 31;
  int tnc = N >> 4;
  int total = (M >> 4) * tnc;
  if (wv >= total) return;                    // wave-uniform: EXEC stays full for WMMA
  int tm = (wv / tnc) << 4;
  int tn = (wv % tnc) << 4;
  v8f acc = {};
  for (int k0 = 0; k0 < K; k0 += 32) {
    v16bf a = load_a(X + (size_t)tm * K, K, k0, lane);
    v16bf b = load_bT(W, K, tn, k0, lane);
    acc = WMMA_BF16(a, b, acc);
  }
  int col = lane & 15, hi = lane >> 4;
  float bv = bias ? bias[tn + col] : 0.f;
#pragma unroll
  for (int r = 0; r < 8; ++r) {
    float v = acc[r] + bv;
    if (act == 1) v = tanhf(v);
    size_t idx = (size_t)(tm + (hi << 3) + r) * N + tn + col;
    if (Yf)  Yf[idx]  = v;
    if (Ybf) Ybf[idx] = (__bf16)v;
  }
}

// ---------------- persistent bidirectional encoder LSTM ----------------
// grid.x = 2 (dir), 1024 threads = 32 waves. Wave w owns hidden cols [16w,16w+16);
// gate g lives in tile (w + 32g) so i/f/g/o line up per lane -> no gate staging LDS.

__global__ __launch_bounds__(1024,1) void k_enc_lstm(
    const float* __restrict__ gxf, const float* __restrict__ gxb,
    const __bf16* __restrict__ whhf, const __bf16* __restrict__ whhb,
    const int* __restrict__ slen,
    __bf16* __restrict__ cat, __bf16* __restrict__ hcat, __bf16* __restrict__ ccat)
{
  const int dir = blockIdx.x;
  const float*  gx  = dir ? gxb  : gxf;
  const __bf16* whh = dir ? whhb : whhf;
  __shared__ __align__(16) __bf16 hbf[CB*CH];
  __shared__ __align__(16) float  cf[CB*CH];
  __shared__ int sl[CB];
  int tid = threadIdx.x;
  int w = tid >> 5, lane = tid & 31;
  for (int i = tid; i < CB*CH; i += 1024){ hbf[i] = (__bf16)0.f; cf[i] = 0.f; }
  if (tid < CB) sl[tid] = slen[tid];
  __syncthreads();
  int col = lane & 15, hi = lane >> 4;
  int n = (w << 4) + col;
  for (int step = 0; step < CS; ++step) {
    int s_act = dir ? (CS - 1 - step) : step;
    v8f acc[4] = {};
    for (int k0 = 0; k0 < CH; k0 += 32) {
      v16bf a = load_a(hbf, CH, k0, lane);
#pragma unroll
      for (int g = 0; g < 4; ++g) {
        v16bf bb = load_bT(whh, CH, (w << 4) + g*CH, k0, lane);
        acc[g] = WMMA_BF16(a, bb, acc[g]);
      }
    }
    __syncthreads();                                   // all A reads of hbf done
#pragma unroll
    for (int r = 0; r < 8; ++r) {
      int b = r + (hi << 3);
      size_t gb = ((size_t)(s_act*CB + b))*G4H + n;    // gx already includes bias
      float gi = acc[0][r] + gx[gb];
      float gf = acc[1][r] + gx[gb + CH];
      float gg = acc[2][r] + gx[gb + 2*CH];
      float go = acc[3][r] + gx[gb + 3*CH];
      float cold = cf[b*CH + n];
      float cn = sigm(gf)*cold + sigm(gi)*tanhf(gg);
      float hn = sigm(go)*tanhf(cn);
      bool mm = s_act < sl[b];
      if (mm) { cf[b*CH + n] = cn; hbf[b*CH + n] = (__bf16)hn; }
      cat[((size_t)b*CS + s_act)*(2*CH) + dir*CH + n] = (__bf16)(mm ? hn : 0.f);
    }
    __syncthreads();                                   // hbf updated before next GEMM
  }
#pragma unroll
  for (int r = 0; r < 8; ++r) {
    int b = r + (hi << 3);
    hcat[b*(2*CH) + dir*CH + n] = hbf[b*CH + n];
    ccat[b*(2*CH) + dir*CH + n] = (__bf16)cf[b*CH + n];
  }
}

// ---------------- misc decoder-init kernels ----------------

__global__ void k_seqmean(const float* __restrict__ eo, const int* __restrict__ slen,
                          float* __restrict__ sm){
  int i = blockIdx.x*256 + threadIdx.x;
  if (i >= CB*CH) return;
  int b = i >> 9, h = i & (CH-1);
  int L = slen[b];
  float s = 0.f;
  for (int ss = 0; ss < L; ++ss) s += eo[((size_t)b*CS + ss)*CH + h];
  sm[i] = s / (float)L;
}

__global__ void k_build_dhsm(const float* __restrict__ dh, const float* __restrict__ sm,
                             __bf16* __restrict__ out){
  int i = blockIdx.x*256 + threadIdx.x;
  if (i >= CB*2*CH) return;
  int b = i >> 10, c = i & (2*CH-1);
  out[i] = (__bf16)((c < CH) ? dh[b*CH + c] : sm[b*CH + c - CH]);
}

// ---------------- fused decoder cell + attention + out_proj ----------------

__global__ __launch_bounds__(1024,1) void k_dec_cell(
    int t,
    const float* __restrict__ wt,
    const __bf16* __restrict__ dwih, const __bf16* __restrict__ dwhh,
    const float* __restrict__ decb,
    __bf16* __restrict__ hstate, float* __restrict__ cstate,
    const float* __restrict__ encouts, const int* __restrict__ slen,
    const float* __restrict__ wpgen, const float* __restrict__ bpgen,
    const __bf16* __restrict__ wo1, const float* __restrict__ bo1,
    const __bf16* __restrict__ wo2,
    float* __restrict__ attn_g, float* __restrict__ pgen_g,
    float* __restrict__ decout_f, __bf16* __restrict__ decout_bf)
{
  __shared__ __align__(16) __bf16 xbf[CB*CH];    // x = [w_t, dec_out] (2D = 512)
  __shared__ __align__(16) __bf16 hbf[CB*CH];
  __shared__ __align__(16) float  hf[CB*CH];
  __shared__ __align__(16) float  sc[CB*CS];     // scores -> attn (in place)
  __shared__ __align__(16) __bf16 ctxbf[CB*CH];
  __shared__ __align__(16) __bf16 z1bf[CB*CH];
  __shared__ int sl[CB];
  int tid = threadIdx.x;
  int w = tid >> 5, lane = tid & 31;
  for (int i = tid; i < CB*CH; i += 1024) {
    int b = i >> 9, c = i & (CH-1);
    float xv = (c < CD) ? wt[((size_t)t*CB + b)*CD + c] : decout_f[b*CD + c - CD];
    xbf[i] = (__bf16)xv;
    hbf[i] = hstate[i];
  }
  if (tid < CB) sl[tid] = slen[tid];
  __syncthreads();
  int col = lane & 15, hi = lane >> 4;
  int n = (w << 4) + col;
  // ---- gates: g = x@Wih.T + h@Whh.T ----
  v8f acc[4] = {};
  for (int k0 = 0; k0 < CH; k0 += 32) {
    v16bf ax = load_a(xbf, CH, k0, lane);
    v16bf ah = load_a(hbf, CH, k0, lane);
#pragma unroll
    for (int g = 0; g < 4; ++g) {
      v16bf b0 = load_bT(dwih, 2*CD, (w << 4) + g*CH, k0, lane);
      acc[g] = WMMA_BF16(ax, b0, acc[g]);
      v16bf b1 = load_bT(dwhh, CH,   (w << 4) + g*CH, k0, lane);
      acc[g] = WMMA_BF16(ah, b1, acc[g]);
    }
  }
  __syncthreads();
#pragma unroll
  for (int r = 0; r < 8; ++r) {
    int b = r + (hi << 3);
    float gi = acc[0][r] + decb[n];
    float gf = acc[1][r] + decb[CH + n];
    float gg = acc[2][r] + decb[2*CH + n];
    float go = acc[3][r] + decb[3*CH + n];
    float cn = sigm(gf)*cstate[b*CH + n] + sigm(gi)*tanhf(gg);
    float hn = sigm(go)*tanhf(cn);
    cstate[b*CH + n] = cn;
    hf[b*CH + n] = hn;
    __bf16 hb = (__bf16)hn;
    hbf[b*CH + n] = hb;
    hstate[b*CH + n] = hb;
  }
  __syncthreads();
  // ---- attention scores: wave w -> batch w>>1, s-half w&1 ----
  {
    int b = w >> 1, s0 = (w & 1) * (CS/2);
    for (int si = 0; si < CS/2; ++si) {
      int s = s0 + si;
      const float* e = encouts + ((size_t)b*CS + s)*CH;
      float p = 0.f;
      int kb = lane * 16;
#pragma unroll
      for (int k = 0; k < 16; ++k) p += hf[b*CH + kb + k] * e[kb + k];
      for (int off = 16; off; off >>= 1) p += __shfl_xor(p, off, 32);
      if (lane == 0) sc[b*CS + s] = p;
    }
  }
  __syncthreads();
  // ---- masked softmax per batch (waves 0..15) ----
  if (w < CB) {
    int b = w, L = sl[b];
    float mx = -3.4e38f;
    for (int s = lane; s < L; s += 32) mx = fmaxf(mx, sc[b*CS + s]);
    for (int off = 16; off; off >>= 1) mx = fmaxf(mx, __shfl_xor(mx, off, 32));
    float sum = 0.f;
    for (int s = lane; s < L; s += 32) sum += expf(sc[b*CS + s] - mx);
    for (int off = 16; off; off >>= 1) sum += __shfl_xor(sum, off, 32);
    float inv = 1.f / sum;
    for (int s = lane; s < CS; s += 32) {
      float a = (s < L) ? expf(sc[b*CS + s] - mx) * inv : 0.f;
      sc[b*CS + s] = a;
      attn_g[b*CS + s] = a;
    }
  }
  __syncthreads();
  // ---- context ----
  for (int i = tid; i < CB*CH; i += 1024) {
    int b = i >> 9, nn = i & (CH-1);
    const float* e = encouts + (size_t)b*CS*CH + nn;
    float s = 0.f;
    for (int ss = 0; ss < CS; ++ss) s += sc[b*CS + ss] * e[(size_t)ss*CH];
    ctxbf[i] = (__bf16)s;
  }
  __syncthreads();
  // ---- p_gen = sigmoid([context, h, x] . Wpgen + b) ----
  if (w < CB) {
    int b = w;
    float p = 0.f;
    for (int k = lane; k < 2*CH + 2*CD; k += 32) {
      float v;
      if (k < CH)           v = (float)ctxbf[b*CH + k];
      else if (k < 2*CH)    v = hf[b*CH + k - CH];
      else                  v = (float)xbf[b*CH + k - 2*CH];
      p += v * wpgen[k];
    }
    for (int off = 16; off; off >>= 1) p += __shfl_xor(p, off, 32);
    if (lane == 0) pgen_g[b] = sigm(p + bpgen[0]);
  }
  __syncthreads();
  // ---- z1 = tanh([h, context] @ Wo1.T + bo1) : 32 tiles, K split 512+512 ----
  {
    v8f a1 = {};
    for (int k0 = 0; k0 < CH; k0 += 32) {
      v16bf ah = load_a(hbf, CH, k0, lane);
      v16bf bb = load_bT(wo1, 2*CH, w << 4, k0, lane);
      a1 = WMMA_BF16(ah, bb, a1);
    }
    for (int k0 = 0; k0 < CH; k0 += 32) {
      v16bf ac = load_a(ctxbf, CH, k0, lane);
      v16bf bb = load_bT(wo1, 2*CH, w << 4, CH + k0, lane);
      a1 = WMMA_BF16(ac, bb, a1);
    }
    float bv = bo1[n];
#pragma unroll
    for (int r = 0; r < 8; ++r) {
      int b = r + (hi << 3);
      z1bf[b*CH + n] = (__bf16)tanhf(a1[r] + bv);
    }
  }
  __syncthreads();
  // ---- dec_out = z1 @ Wo2.T : 16 tiles (waves 0..15) ----
  if (w < CB) {
    v8f a2 = {};
    for (int k0 = 0; k0 < CH; k0 += 32) {
      v16bf az = load_a(z1bf, CH, k0, lane);
      v16bf bb = load_bT(wo2, CH, w << 4, k0, lane);
      a2 = WMMA_BF16(az, bb, a2);
    }
#pragma unroll
    for (int r = 0; r < 8; ++r) {
      int b = r + (hi << 3);
      float v = a2[r];
      decout_f[b*CD + n]  = v;
      decout_bf[b*CD + n] = (__bf16)v;
    }
  }
}

// ---------------- vocab softmax / pointer mix / log ----------------

__global__ void k_softstats(const float* __restrict__ logit, float* __restrict__ smax,
                            float* __restrict__ ssum){
  int b = blockIdx.x, tid = threadIdx.x;
  __shared__ float red[256];
  float mx = -3.4e38f;
  for (int v = tid; v < CV; v += 256) mx = fmaxf(mx, logit[(size_t)b*CV + v]);
  red[tid] = mx; __syncthreads();
  for (int s = 128; s; s >>= 1){ if (tid < s) red[tid] = fmaxf(red[tid], red[tid+s]); __syncthreads(); }
  mx = red[0]; __syncthreads();
  float sm = 0.f;
  for (int v = tid; v < CV; v += 256) sm += expf(logit[(size_t)b*CV + v] - mx);
  red[tid] = sm; __syncthreads();
  for (int s = 128; s; s >>= 1){ if (tid < s) red[tid] += red[tid+s]; __syncthreads(); }
  if (tid == 0){ smax[b] = mx; ssum[b] = red[0]; }
}

__global__ void k_vocab(const float* __restrict__ logit, const float* __restrict__ smax,
                        const float* __restrict__ ssum, const float* __restrict__ pg,
                        float* __restrict__ outt){
  int idx = blockIdx.x*256 + threadIdx.x;
  if (idx >= CB*VE) return;
  int b = idx / VE, v = idx - b*VE;
  float val = 0.f;
  if (v < CV) val = pg[b] * expf(logit[(size_t)b*CV + v] - smax[b]) / ssum[b];
  outt[idx] = val;
}

__global__ void k_scatter(const int* __restrict__ se, const float* __restrict__ attn,
                          const float* __restrict__ pg, float* __restrict__ outt){
  for (int i = threadIdx.x; i < CB*CS; i += blockDim.x) {
    int b = i / CS;
    float a = attn[i];
    if (a != 0.f) atomicAdd(&outt[(size_t)b*VE + se[i]], (1.f - pg[b]) * a);
  }
}

__global__ void k_log(float* __restrict__ outt, int n){
  int i = blockIdx.x*256 + threadIdx.x;
  if (i < n) outt[i] = logf(outt[i] + 1e-20f);
}

// ---------------- host ----------------

extern "C" void kernel_launch(void* const* d_in, const int* in_sizes, int n_in,
                              void* d_out, int out_size, void* d_ws, size_t ws_size,
                              hipStream_t stream) {
  (void)in_sizes; (void)n_in; (void)out_size; (void)ws_size;
  const int*   src    = (const int*)d_in[0];
  const int*   srcext = (const int*)d_in[1];
  const int*   slen   = (const int*)d_in[2];
  const int*   tgt    = (const int*)d_in[3];
  const float* emb    = (const float*)d_in[4];
  const float* eWihF  = (const float*)d_in[5];
  const float* eWhhF  = (const float*)d_in[6];
  const float* ebF    = (const float*)d_in[7];
  const float* eWihB  = (const float*)d_in[8];
  const float* eWhhB  = (const float*)d_in[9];
  const float* ebB    = (const float*)d_in[10];
  const float* Wproj  = (const float*)d_in[11];
  const float* bproj  = (const float*)d_in[12];
  const float* W2h    = (const float*)d_in[13];
  const float* b2h    = (const float*)d_in[14];
  const float* W2c    = (const float*)d_in[15];
  const float* b2c    = (const float*)d_in[16];
  const float* Wo1    = (const float*)d_in[17];
  const float* bo1    = (const float*)d_in[18];
  const float* Wo2    = (const float*)d_in[19];
  const float* dWih   = (const float*)d_in[20];
  const float* dWhh   = (const float*)d_in[21];
  const float* db     = (const float*)d_in[22];
  const float* Wpg    = (const float*)d_in[23];
  const float* bpg    = (const float*)d_in[24];

  char* p = (char*)d_ws;
  auto alloc = [&](size_t bytes)->char*{
    char* r = p; p += (bytes + 255) & ~(size_t)255; return r;
  };
  __bf16* emb_bf   = (__bf16*)alloc((size_t)CV*CD*2);
  __bf16* wihf_bf  = (__bf16*)alloc((size_t)G4H*CD*2);
  __bf16* whhf_bf  = (__bf16*)alloc((size_t)G4H*CH*2);
  __bf16* wihb_bf  = (__bf16*)alloc((size_t)G4H*CD*2);
  __bf16* whhb_bf  = (__bf16*)alloc((size_t)G4H*CH*2);
  __bf16* wproj_bf = (__bf16*)alloc((size_t)CH*2*CH*2);
  __bf16* w2h_bf   = (__bf16*)alloc((size_t)CH*2*CH*2);
  __bf16* w2c_bf   = (__bf16*)alloc((size_t)CH*2*CH*2);
  __bf16* wo1_bf   = (__bf16*)alloc((size_t)CH*2*CH*2);
  __bf16* wo2_bf   = (__bf16*)alloc((size_t)CD*CH*2);
  __bf16* dwih_bf  = (__bf16*)alloc((size_t)G4H*2*CD*2);
  __bf16* dwhh_bf  = (__bf16*)alloc((size_t)G4H*CH*2);
  __bf16* xs_bf    = (__bf16*)alloc((size_t)CS*CB*CD*2);
  __bf16* cat_bf   = (__bf16*)alloc((size_t)CB*CS*2*CH*2);
  __bf16* hcat_bf  = (__bf16*)alloc((size_t)CB*2*CH*2);
  __bf16* ccat_bf  = (__bf16*)alloc((size_t)CB*2*CH*2);
  __bf16* hstate   = (__bf16*)alloc((size_t)CB*CH*2);
  __bf16* dhsm_bf  = (__bf16*)alloc((size_t)CB*2*CH*2);
  __bf16* z1_bf    = (__bf16*)alloc((size_t)CB*CH*2);
  __bf16* dout_bf  = (__bf16*)alloc((size_t)CB*CD*2);
  float* wt        = (float*)alloc((size_t)CT*CB*CD*4);
  float* gxf       = (float*)alloc((size_t)CS*CB*G4H*4);
  float* gxb       = (float*)alloc((size_t)CS*CB*G4H*4);
  float* encouts   = (float*)alloc((size_t)CB*CS*CH*4);
  float* dech0     = (float*)alloc((size_t)CB*CH*4);
  float* cstate    = (float*)alloc((size_t)CB*CH*4);
  float* seqmean   = (float*)alloc((size_t)CB*CH*4);
  float* dout_f    = (float*)alloc((size_t)CB*CD*4);
  float* attn      = (float*)alloc((size_t)CB*CS*4);
  float* pgen      = (float*)alloc((size_t)CB*4);
  float* smax      = (float*)alloc((size_t)CB*4);
  float* ssum      = (float*)alloc((size_t)CB*4);
  float* logit     = (float*)alloc((size_t)CB*CV*4);

  auto cvt = [&](const float* s, __bf16* d, long n){
    k_f2b<<<(int)((n + 255)/256), 256, 0, stream>>>(s, d, n);
  };
  cvt(emb,   emb_bf,   (long)CV*CD);
  cvt(eWihF, wihf_bf,  (long)G4H*CD);
  cvt(eWhhF, whhf_bf,  (long)G4H*CH);
  cvt(eWihB, wihb_bf,  (long)G4H*CD);
  cvt(eWhhB, whhb_bf,  (long)G4H*CH);
  cvt(Wproj, wproj_bf, (long)CH*2*CH);
  cvt(W2h,   w2h_bf,   (long)CH*2*CH);
  cvt(W2c,   w2c_bf,   (long)CH*2*CH);
  cvt(Wo1,   wo1_bf,   (long)CH*2*CH);
  cvt(Wo2,   wo2_bf,   (long)CD*CH);
  cvt(dWih,  dwih_bf,  (long)G4H*2*CD);
  cvt(dWhh,  dwhh_bf,  (long)G4H*CH);

  k_embed_src<<<(int)(((long)CS*CB*CD + 255)/256), 256, 0, stream>>>(src, emb, xs_bf);
  k_embed_tgt<<<(int)(((long)CT*CB*CD + 255)/256), 256, 0, stream>>>(tgt, emb, wt);

  auto gemm = [&](const __bf16* X, const __bf16* W, const float* bias,
                  float* Yf, __bf16* Ybf, int M, int N, int K, int act){
    int tiles = (M >> 4) * (N >> 4);
    k_gemm_xt<<<(tiles + 7)/8, 256, 0, stream>>>(X, W, bias, Yf, Ybf, M, N, K, act);
  };

  // encoder input-gate precompute (bias folded in)
  gemm(xs_bf, wihf_bf, ebF, gxf, nullptr, CS*CB, G4H, CD, 0);
  gemm(xs_bf, wihb_bf, ebB, gxb, nullptr, CS*CB, G4H, CD, 0);
  // recurrent bidirectional LSTM (2 persistent blocks)
  k_enc_lstm<<<2, 1024, 0, stream>>>(gxf, gxb, whhf_bf, whhb_bf, slen,
                                     cat_bf, hcat_bf, ccat_bf);
  // projection + decoder init
  gemm(cat_bf, wproj_bf, bproj, encouts, nullptr, CB*CS, CH, 2*CH, 0);
  k_seqmean<<<(CB*CH + 255)/256, 256, 0, stream>>>(encouts, slen, seqmean);
  gemm(hcat_bf, w2h_bf, b2h, dech0, hstate, CB, CH, 2*CH, 0);
  gemm(ccat_bf, w2c_bf, b2c, cstate, nullptr, CB, CH, 2*CH, 0);
  k_build_dhsm<<<(CB*2*CH + 255)/256, 256, 0, stream>>>(dech0, seqmean, dhsm_bf);
  gemm(dhsm_bf, wo1_bf, bo1, nullptr, z1_bf, CB, CH, 2*CH, 1);
  gemm(z1_bf, wo2_bf, nullptr, dout_f, dout_bf, CB, CD, CH, 0);

  // decoder loop
  const int nout = CB*VE;
  for (int t = 0; t < CT; ++t) {
    k_dec_cell<<<1, 1024, 0, stream>>>(t, wt, dwih_bf, dwhh_bf, db,
                                       hstate, cstate, encouts, slen,
                                       Wpg, bpg, wo1_bf, bo1, wo2_bf,
                                       attn, pgen, dout_f, dout_bf);
    gemm(dout_bf, emb_bf, nullptr, logit, nullptr, CB, CV, CD, 0);
    k_softstats<<<CB, 256, 0, stream>>>(logit, smax, ssum);
    float* outt = (float*)d_out + (size_t)t*nout;
    k_vocab<<<(nout + 255)/256, 256, 0, stream>>>(logit, smax, ssum, pgen, outt);
    k_scatter<<<1, 1024, 0, stream>>>(srcext, attn, pgen, outt);
    k_log<<<(nout + 255)/256, 256, 0, stream>>>(outt, nout);
  }
}
